// EncSwitchedFC_44985487458666
// MI455X (gfx1250) — compile-verified, matrix operand order checked by reading, not run on hardware
//
#include <hip/hip_runtime.h>

#define B_ROWS 131072
#define D_DIM  256
#define H_DIM  64
#define NBE    8

typedef __attribute__((ext_vector_type(16))) __bf16        v16bf;
typedef __attribute__((ext_vector_type(8)))  float         v8f;
typedef __attribute__((ext_vector_type(4)))  unsigned int  v4u;

union FragU { v16bf v; struct { v4u lo, hi; } q; };

// Load a 16x32 bf16 WMMA operand fragment from an LDS tile.
// ISA layout (CDNA5 ISA): lanes 0-15 hold row (lane), K=0..7 then K=16..23;
// lanes 16-31 hold row (lane-16), K=8..15 then K=24..31.
// 'tile' points at element (row0, kbase); stride in elements; 16B aligned.
__device__ inline v16bf load_frag(const __bf16* tile, int stride) {
  const int lane = threadIdx.x & 31;
  const int r    = lane & 15;
  const int koff = (lane & 16) ? 8 : 0;
  const __bf16* p = tile + r * stride + koff;
  FragU f;
  f.q.lo = *(const v4u*)(p);
  f.q.hi = *(const v4u*)(p + 16);
  return f.v;
}

__device__ inline v8f wmma_bf16(v16bf a, v16bf b, v8f c) {
  return __builtin_amdgcn_wmma_f32_16x16x32_bf16(false, a, false, b, (short)0, c,
                                                 false, false);
}

__global__ __launch_bounds__(32) void k_zero(int* __restrict__ counts) {
  if (threadIdx.x < NBE) counts[threadIdx.x] = 0;
}

// ---------------------------------------------------------------------------
// Kernel 1: routing GEMM (relu(x) @ W_sw) via WMMA + per-row gumbel routing.
// Block = 256 threads (8 waves), 128 rows. Also bins rows by chosen expert.
// ---------------------------------------------------------------------------
__global__ __launch_bounds__(256) void k_route(
    const float* __restrict__ x, const float* __restrict__ W_sw,
    const float* __restrict__ b_sw, const float* __restrict__ exp_noise,
    const float* __restrict__ gauss_noise,
    float* __restrict__ out_ylogits, float* __restrict__ out_yindex,
    float* __restrict__ out_yhard, float* __restrict__ out_zmean,
    float* __restrict__ out_zlogvar, float* __restrict__ out_zs,
    int* __restrict__ counts, int* __restrict__ bins, float* __restrict__ zsel)
{
  __shared__ __align__(16) __bf16 sA[128][264];  // relu(x) tile, bf16
  __shared__ __align__(16) __bf16 sW[32][264];   // W_sw^T padded to 32 cols
  __shared__ float sC[8][16][33];                // per-wave ctrl results

  const int  tid     = threadIdx.x;
  const int  wave    = tid >> 5;
  const long rowBase = (long)blockIdx.x * 128;

  // Stage W_sw transposed (n-major, k contiguous) as bf16, zero-pad n>=24.
  for (int i = tid; i < 32 * 256; i += 256) {
    const int n = i & 31, k = i >> 5;
    const float w = (n < 24) ? W_sw[k * 24 + n] : 0.0f;
    sW[n][k] = (__bf16)w;
  }
  // Stage relu(x) as bf16 (coalesced along K).
  for (int i = tid; i < 128 * 256; i += 256) {
    const int r = i >> 8, c = i & 255;
    const float v = x[(rowBase + r) * 256 + c];
    sA[r][c] = (__bf16)fmaxf(v, 0.0f);
  }
  __syncthreads();

  // ctrl tile: 16 rows x 32 cols (24 live), K=256 in 8 steps of 32.
  v8f c0 = {}, c1 = {};
#pragma unroll
  for (int k = 0; k < 8; ++k) {
    const v16bf a  = load_frag(&sA[wave * 16][k * 32], 264);
    const v16bf b0 = load_frag(&sW[0][k * 32], 264);
    const v16bf b1 = load_frag(&sW[16][k * 32], 264);
    c0 = wmma_bf16(a, b0, c0);
    c1 = wmma_bf16(a, b1, c1);
  }
  {
    const int lane = tid & 31;
    const int n    = lane & 15;
    const int mB   = (lane & 16) ? 8 : 0;
#pragma unroll
    for (int v = 0; v < 8; ++v) {
      sC[wave][mB + v][n]      = c0[v];
      sC[wave][mB + v][16 + n] = c1[v];
    }
  }
  __syncthreads();

  // Per-row routing math: one thread per row.
  if (tid < 128) {
    const int  r = tid;
    const long b = rowBase + r;
    const int  w = r >> 4, m = r & 15;

    float yl[NBE], zm[NBE], zv[NBE];
#pragma unroll
    for (int j = 0; j < NBE; ++j) {
      yl[j] = sC[w][m][j]      + b_sw[j];
      zm[j] = sC[w][m][8 + j]  + b_sw[8 + j];
      zv[j] = sC[w][m][16 + j] + b_sw[16 + j];
    }
    float g[NBE], gmax = -1e30f;
#pragma unroll
    for (int j = 0; j < NBE; ++j) {
      g[j] = yl[j] - __logf(exp_noise[b * NBE + j] + 1e-20f);
      gmax = fmaxf(gmax, g[j]);
    }
    float ys[NBE], esum = 0.0f;
#pragma unroll
    for (int j = 0; j < NBE; ++j) { ys[j] = __expf(g[j] - gmax); esum += ys[j]; }
    const float inv = 1.0f / esum;
    int idx = 0; float best = -1.0f;
#pragma unroll
    for (int j = 0; j < NBE; ++j) {
      ys[j] *= inv;
      if (ys[j] > best) { best = ys[j]; idx = j; }   // first-max, like argmax
    }
    float z[NBE];
#pragma unroll
    for (int j = 0; j < NBE; ++j) {
      z[j] = gauss_noise[b * NBE + j] * __expf(0.5f * zv[j]) + zm[j];
      out_ylogits[b * NBE + j] = yl[j];
      const float oh = (j == idx) ? 1.0f : 0.0f;
      out_yhard[b * NBE + j] = (oh - ys[j]) + ys[j];  // as written in reference
    }
    out_yindex[b]  = (float)idx;
    out_zmean[b]   = zm[idx];
    out_zlogvar[b] = zv[idx];
    out_zs[b]      = z[idx];
    zsel[b]        = z[idx];

    const int pos = atomicAdd(&counts[idx], 1);
    bins[(long)idx * B_ROWS + pos] = (int)b;
  }
}

// ---------------------------------------------------------------------------
// Kernel 2: per-expert MLP on binned rows. grid = (8 experts, 1024 chunks).
// Block = 256 threads (8 waves), <=128 gathered rows. All GEMMs via WMMA.
// ---------------------------------------------------------------------------
__global__ __launch_bounds__(256) void k_expert(
    const float* __restrict__ x,
    const float* __restrict__ W1, const float* __restrict__ b1,
    const float* __restrict__ W2, const float* __restrict__ b2,
    const int* __restrict__ counts, const int* __restrict__ bins,
    const float* __restrict__ zsel, float* __restrict__ out)
{
  __shared__ __align__(16) __bf16 sA[128][264];    // relu(x[rows])
  __shared__ __align__(16) __bf16 sW1[64][264];    // W1[e]^T  [n][k], k<256
  __shared__ __align__(16) __bf16 sW2[256][72];    // W2[e]^T  [n][k], k<64
  __shared__ __align__(16) __bf16 sH[8][16][72];   // per-wave h tiles (bf16)
  __shared__ float sB1[64];
  __shared__ float sB2[256];
  __shared__ int   sRows[128];
  __shared__ float sZ[128];

  const int e     = blockIdx.x;
  const int tid   = threadIdx.x;
  const int cnt   = counts[e];
  const int start = (int)blockIdx.y * 128;
  if (start >= cnt) return;                 // uniform across block
  const int nrows = (cnt - start < 128) ? (cnt - start) : 128;

  if (tid < 128) {
    const int gr = (tid < nrows) ? bins[(long)e * B_ROWS + start + tid] : 0;
    sRows[tid] = gr;
    sZ[tid]    = (tid < nrows) ? zsel[gr] : 0.0f;
  }
  if (tid < 64) sB1[tid] = b1[e * 64 + tid];
  sB2[tid] = b2[e * 256 + tid];
  __syncthreads();

  // Stage W1^T and W2^T (coalesced global reads, transposed LDS writes).
  for (int i = tid; i < 64 * 256; i += 256) {
    const int n = i & 63, k = i >> 6;
    sW1[n][k] = (__bf16)W1[((long)e * 256 + k) * 64 + n];
  }
  for (int i = tid; i < 64 * 256; i += 256) {
    const int n = i & 255, k = i >> 8;
    sW2[n][k] = (__bf16)W2[((long)e * 64 + k) * 256 + n];
  }
  // Stage gathered relu(x) rows as bf16; x lives in the 192MB L2 after k_route.
  for (int i = tid; i < 128 * 256; i += 256) {
    const int r = i >> 8, c = i & 255;
    const int rn = (i + 256) >> 8;
    if (rn < nrows)
      __builtin_prefetch(&x[(long)sRows[rn] * 256 + ((i + 256) & 255)], 0, 3);
    const float v = (r < nrows) ? x[(long)sRows[r] * 256 + c] : 0.0f;
    sA[r][c] = (__bf16)fmaxf(v, 0.0f);
  }
  __syncthreads();

  const int wave = tid >> 5;
  const int lane = tid & 31;

  // Stage 1: h = relu(A @ W1 + b1), per-wave 16x64 tile, K=256 (8 steps).
  v8f h0 = {}, h1 = {}, h2 = {}, h3 = {};
#pragma unroll
  for (int k = 0; k < 8; ++k) {
    const v16bf a = load_frag(&sA[wave * 16][k * 32], 264);
    h0 = wmma_bf16(a, load_frag(&sW1[0][k * 32], 264), h0);
    h1 = wmma_bf16(a, load_frag(&sW1[16][k * 32], 264), h1);
    h2 = wmma_bf16(a, load_frag(&sW1[32][k * 32], 264), h2);
    h3 = wmma_bf16(a, load_frag(&sW1[48][k * 32], 264), h3);
  }
  {
    const int n  = lane & 15;
    const int mB = (lane & 16) ? 8 : 0;
#pragma unroll
    for (int v = 0; v < 8; ++v) {
      sH[wave][mB + v][n]      = (__bf16)fmaxf(h0[v] + sB1[n],      0.0f);
      sH[wave][mB + v][16 + n] = (__bf16)fmaxf(h1[v] + sB1[16 + n], 0.0f);
      sH[wave][mB + v][32 + n] = (__bf16)fmaxf(h2[v] + sB1[32 + n], 0.0f);
      sH[wave][mB + v][48 + n] = (__bf16)fmaxf(h3[v] + sB1[48 + n], 0.0f);
    }
  }
  __syncthreads();

  // Stage 2: o = h @ W2 + b2; out = x + o * z_sel. 16 N-tiles in 4 groups.
#pragma unroll
  for (int g = 0; g < 4; ++g) {
    v8f o0 = {}, o1 = {}, o2 = {}, o3 = {};
#pragma unroll
    for (int k = 0; k < 2; ++k) {
      const v16bf a = load_frag(&sH[wave][0][k * 32], 72);
      o0 = wmma_bf16(a, load_frag(&sW2[(g * 4 + 0) * 16][k * 32], 72), o0);
      o1 = wmma_bf16(a, load_frag(&sW2[(g * 4 + 1) * 16][k * 32], 72), o1);
      o2 = wmma_bf16(a, load_frag(&sW2[(g * 4 + 2) * 16][k * 32], 72), o2);
      o3 = wmma_bf16(a, load_frag(&sW2[(g * 4 + 3) * 16][k * 32], 72), o3);
    }
    const int n0 = lane & 15;
    const int mB = (lane & 16) ? 8 : 0;
#pragma unroll
    for (int v = 0; v < 8; ++v) {
      const int r = wave * 16 + mB + v;
      if (r < nrows) {
        const long  gr = sRows[r];
        const float zz = sZ[r];
        {
          const int n = (g * 4 + 0) * 16 + n0;
          out[gr * 256 + n] = x[gr * 256 + n] + (o0[v] + sB2[n]) * zz;
        }
        {
          const int n = (g * 4 + 1) * 16 + n0;
          out[gr * 256 + n] = x[gr * 256 + n] + (o1[v] + sB2[n]) * zz;
        }
        {
          const int n = (g * 4 + 2) * 16 + n0;
          out[gr * 256 + n] = x[gr * 256 + n] + (o2[v] + sB2[n]) * zz;
        }
        {
          const int n = (g * 4 + 3) * 16 + n0;
          out[gr * 256 + n] = x[gr * 256 + n] + (o3[v] + sB2[n]) * zz;
        }
      }
    }
  }
}

extern "C" void kernel_launch(void* const* d_in, const int* in_sizes, int n_in,
                              void* d_out, int out_size, void* d_ws, size_t ws_size,
                              hipStream_t stream) {
  (void)in_sizes; (void)n_in; (void)out_size; (void)ws_size;

  const float* x           = (const float*)d_in[0];
  const float* W_sw        = (const float*)d_in[1];
  const float* b_sw        = (const float*)d_in[2];
  const float* W1          = (const float*)d_in[3];
  const float* b1          = (const float*)d_in[4];
  const float* W2          = (const float*)d_in[5];
  const float* b2          = (const float*)d_in[6];
  const float* exp_noise   = (const float*)d_in[7];
  const float* gauss_noise = (const float*)d_in[8];

  // Outputs concatenated flat in reference return order (y_index stored as
  // its float value since the harness views non-bf16 outputs as float).
  float* out         = (float*)d_out;
  float* out_sel     = out;                                    // B*256
  float* out_ylogits = out_sel + (size_t)B_ROWS * 256;         // B*8
  float* out_yindex  = out_ylogits + (size_t)B_ROWS * NBE;     // B
  float* out_yhard   = out_yindex + (size_t)B_ROWS;            // B*8
  float* out_zmean   = out_yhard + (size_t)B_ROWS * NBE;       // B
  float* out_zlogvar = out_zmean + (size_t)B_ROWS;             // B
  float* out_zs      = out_zlogvar + (size_t)B_ROWS;           // B

  char*  ws     = (char*)d_ws;
  int*   counts = (int*)ws;                                    // 8 ints
  int*   bins   = (int*)(ws + 256);                            // 8*B ints
  float* zs     = (float*)(ws + 256 + (size_t)NBE * B_ROWS * 4); // B floats

  k_zero<<<1, 32, 0, stream>>>(counts);
  k_route<<<B_ROWS / 128, 256, 0, stream>>>(
      x, W_sw, b_sw, exp_noise, gauss_noise,
      out_ylogits, out_yindex, out_yhard, out_zmean, out_zlogvar, out_zs,
      counts, bins, zs);
  dim3 grid2(NBE, B_ROWS / 128);
  k_expert<<<grid2, 256, 0, stream>>>(x, W1, b1, W2, b2, counts, bins, zs, out_sel);
}
